// ChunkedPairwiseEmbedder_27848567947693
// MI455X (gfx1250) — matrix operand order repaired and checked by reading, not compiled, not run on hardware
//
#include <hip/hip_runtime.h>
#include <hip/hip_bf16.h>
#include <stdint.h>

#define Dq 4
#define Lq 8192
#define Kq 32
#define Iq 1024
#define CT 128
#define CZ 128
#define CA 64
#define NL 4
#define RMS_EPS 1e-6f

typedef __attribute__((ext_vector_type(16))) __bf16 v16bf;
typedef __attribute__((ext_vector_type(8)))  float  v8f;
typedef unsigned short ushort_t;
typedef unsigned int   uint_t;

union BfVec { uint4 q[2]; v16bf v; };

__device__ __forceinline__ ushort_t f2bf(float x) {
  uint_t u = __float_as_uint(x);
  u += 0x7FFFu + ((u >> 16) & 1u);   // round-to-nearest-even
  return (ushort_t)(u >> 16);
}
__device__ __forceinline__ uint_t pack2(float a, float b) {
  return (uint_t)f2bf(a) | ((uint_t)f2bf(b) << 16);
}
__device__ __forceinline__ int clampi(int v, int lo, int hi) {
  return v < lo ? lo : (v > hi ? hi : v);
}

// A-operand for v_wmma_f32_16x16x32_bf16.
// Lane L (m = L&15, hi = L>>4): halves [kc*32 + hi*8 .. +7] then [+16 .. +23].
__device__ __forceinline__ v16bf load_a(const ushort_t* A, int stride, int row,
                                        int kc, int lane) {
  const int hi = lane >> 4;
  const ushort_t* base = A + row * stride + kc * 32 + hi * 8;
  BfVec u;
  u.q[0] = *(const uint4*)(base);
  u.q[1] = *(const uint4*)(base + 16);
  return u.v;
}
// B-operand from transposed weights WT[N][K]:
// lane n = L&15 (column), hi = L>>4: halves [kc*32 + hi*8 .. +7] then [+16 .. +23].
__device__ __forceinline__ v16bf load_b(const ushort_t* WT, int stride, int col,
                                        int kc, int lane) {
  const int hi = lane >> 4;
  const ushort_t* base = WT + col * stride + kc * 32 + hi * 8;
  BfVec u;
  u.q[0] = *(const uint4*)(base);
  u.q[1] = *(const uint4*)(base + 16);
  return u.v;
}

// ---- weight prep: f32 [K][N] -> bf16 transposed [N][K] in workspace ----
__global__ void prep_weights(const float* __restrict__ Wsm, const float* __restrict__ Wz,
                             const float* __restrict__ W1,  const float* __restrict__ W2,
                             const float* __restrict__ W3,  ushort_t* __restrict__ wt) {
  const int total = CA * CT * 2 + CA * CA * 3;
  for (int i = blockIdx.x * blockDim.x + threadIdx.x; i < total;
       i += gridDim.x * blockDim.x) {
    int j = i;
    if (j < CA * CT) { int n = j / CT, k = j % CT; wt[i] = f2bf(Wsm[k * CA + n]); continue; }
    j -= CA * CT;
    if (j < CA * CT) { int n = j / CT, k = j % CT; wt[i] = f2bf(Wz[k * CA + n]); continue; }
    j -= CA * CT;
    if (j < CA * CA) { int n = j / CA, k = j % CA; wt[i] = f2bf(W1[k * CA + n]); continue; }
    j -= CA * CA;
    if (j < CA * CA) { int n = j / CA, k = j % CA; wt[i] = f2bf(W2[k * CA + n]); continue; }
    j -= CA * CA;
    { int n = j / CA, k = j % CA; wt[i] = f2bf(W3[k * CA + n]); }
  }
}

// ---- fused kernel: one block = (d, 4 consecutive l) ----
__global__ __launch_bounds__(256) void fused_kernel(
    const int* __restrict__ indices, const float* __restrict__ C_L,
    const float* __restrict__ Z, const int* __restrict__ tok_idx,
    const float* __restrict__ motif_pos, const float* __restrict__ ref_pos,
    const int* __restrict__ ref_uid, const unsigned char* __restrict__ is_mc,
    const unsigned char* __restrict__ is_ms,
    const float* __restrict__ Wdm, const float* __restrict__ Wim,
    const float* __restrict__ Wmm, const float* __restrict__ Wdr,
    const float* __restrict__ Wir, const float* __restrict__ Wmr,
    const float* __restrict__ Wsl, const float* __restrict__ rms_w,
    const ushort_t* __restrict__ wt, float* __restrict__ out) {
  __shared__ __align__(16) ushort_t sWTm[CA * CT];   // 16 KB
  __shared__ __align__(16) ushort_t sWTz[CA * CT];   // 16 KB
  __shared__ __align__(16) ushort_t sWT1[CA * CA];   // 8 KB
  __shared__ __align__(16) ushort_t sWT2[CA * CA];   // 8 KB
  __shared__ __align__(16) ushort_t sWT3[CA * CA];   // 8 KB
  __shared__ __align__(16) unsigned char arena[16384]; // A1(8K)|A2(8K), reused
  __shared__ float sGeoW[10 * CA];
  __shared__ float sRow[Kq * 12];
  __shared__ float sRL[CT];
  __shared__ float sRMSW[CZ];
  __shared__ float sPart[4][CA];
  __shared__ int   sVI[Kq];

  ushort_t* sA1   = (ushort_t*)arena;            // phase 1
  ushort_t* sA2   = (ushort_t*)(arena + 8192);   // phase 1
  ushort_t* sH    = (ushort_t*)(arena + 8192);   // MLP staging (after phase 1)
  float*    sPout = (float*)arena;               // output staging (after phase 1)

  const int tid   = threadIdx.x;
  const int lane  = tid & 31;
  const int wave  = tid >> 5;
  const int mtile = wave >> 2;   // 0..1
  const int ntile = wave & 3;    // 0..3
  const int bid   = blockIdx.x;
  const int d     = bid / (Lq / NL);
  const int lbase = (bid % (Lq / NL)) * NL;

  // --- stage weights into LDS (bf16, transposed) ---
  for (int i = tid; i < CA * CT; i += 256) {
    sWTm[i] = wt[i];
    sWTz[i] = wt[CA * CT + i];
  }
  for (int i = tid; i < CA * CA; i += 256) {
    sWT1[i] = wt[2 * CA * CT + i];
    sWT2[i] = wt[2 * CA * CT + CA * CA + i];
    sWT3[i] = wt[2 * CA * CT + 2 * CA * CA + i];
  }
  for (int i = tid; i < 10 * CA; i += 256) {
    const int r = i >> 6, n = i & 63;
    float v;
    switch (r) {
      case 0: v = Wdm[n];        break;
      case 1: v = Wdm[64 + n];   break;
      case 2: v = Wdm[128 + n];  break;
      case 3: v = Wim[n];        break;
      case 4: v = Wmm[n];        break;
      case 5: v = Wdr[n];        break;
      case 6: v = Wdr[64 + n];   break;
      case 7: v = Wdr[128 + n];  break;
      case 8: v = Wir[n];        break;
      default: v = Wmr[n];       break;
    }
    sGeoW[i] = v;
  }
  if (tid < CZ) sRMSW[tid] = rms_w[tid];
  __syncthreads();

  for (int li = 0; li < NL; ++li) {
    const int l = lbase + li;

    // --- per-row geometry scalars + gather indices ---
    if (tid < Kq) {
      const int v = clampi(indices[((size_t)d * Lq + l) * Kq + tid], 0, Lq - 1);
      sVI[tid] = v;
      const float dx = motif_pos[3 * l]     - motif_pos[3 * v];
      const float dy = motif_pos[3 * l + 1] - motif_pos[3 * v + 1];
      const float dz = motif_pos[3 * l + 2] - motif_pos[3 * v + 2];
      const float invm = 1.f / (1.f + dx * dx + dy * dy + dz * dz);
      const float vmm  = is_mc[v] ? 1.f : 0.f;
      const float ex = ref_pos[3 * l]     - ref_pos[3 * v];
      const float ey = ref_pos[3 * l + 1] - ref_pos[3 * v + 1];
      const float ez = ref_pos[3 * l + 2] - ref_pos[3 * v + 2];
      const float invr = 1.f / (1.f + ex * ex + ey * ey + ez * ez);
      const float vmr  = ((ref_uid[v] == ref_uid[l]) && is_ms[v]) ? 1.f : 0.f;
      float* rs = &sRow[tid * 12];
      rs[0] = dx; rs[1] = dy; rs[2] = dz; rs[3] = invm; rs[4] = vmm;
      rs[5] = ex; rs[6] = ey; rs[7] = ez; rs[8] = invr; rs[9] = vmr;
    }
    if (tid < CT) {
      const float c = C_L[((size_t)d * Lq + l) * CT + tid];
      sRL[tid] = c > 0.f ? c : 0.f;
    }
    __syncthreads();

    // --- build A1 = relu(C_keys) and A2 = rmsnorm(Z-row)*rms_w as bf16 ---
    {
      const int k = tid >> 3, seg = tid & 7;
      const int vk = sVI[k];
      // A1: gather C_L[d, vk, seg*16 .. +15], relu, cvt bf16
      const float4* src = (const float4*)(C_L + ((size_t)d * Lq + vk) * CT + seg * 16);
      uint_t* dst = (uint_t*)(sA1 + k * CT + seg * 16);
#pragma unroll
      for (int j = 0; j < 4; ++j) {
        const float4 f = src[j];
        dst[2 * j]     = pack2(fmaxf(f.x, 0.f), fmaxf(f.y, 0.f));
        dst[2 * j + 1] = pack2(fmaxf(f.z, 0.f), fmaxf(f.w, 0.f));
      }
      // A2: gather Z[q, tk, :], rms-normalize across 8 lanes sharing a row
      const int q  = clampi(tok_idx[l], 0, Iq - 1);
      const int tk = clampi(tok_idx[vk], 0, Iq - 1);
      const float4* zs = (const float4*)(Z + ((size_t)q * Iq + tk) * CZ + seg * 16);
      const float4 z0 = zs[0], z1 = zs[1], z2 = zs[2], z3 = zs[3];
      const float e[16] = { z0.x, z0.y, z0.z, z0.w, z1.x, z1.y, z1.z, z1.w,
                            z2.x, z2.y, z2.z, z2.w, z3.x, z3.y, z3.z, z3.w };
      float ss = 0.f;
#pragma unroll
      for (int j = 0; j < 16; ++j) ss += e[j] * e[j];
#pragma unroll
      for (int off = 1; off < 8; off <<= 1) ss += __shfl_xor(ss, off, 32);
      const float sc = rsqrtf(ss * (1.f / (float)CZ) + RMS_EPS);
      uint_t* zdst = (uint_t*)(sA2 + k * CT + seg * 16);
      const float* rw = sRMSW + seg * 16;
#pragma unroll
      for (int j = 0; j < 8; ++j)
        zdst[j] = pack2(e[2 * j] * sc * rw[2 * j], e[2 * j + 1] * sc * rw[2 * j + 1]);
    }
    __syncthreads();

    // --- single_l partial sums (VALU; tiny 1x128 @ 128x64) ---
    {
      const int n = tid & 63, part = tid >> 6;
      float s = 0.f;
      const float* w = Wsl + n;
#pragma unroll 8
      for (int k2 = part * 32; k2 < part * 32 + 32; ++k2)
        s += sRL[k2] * w[(size_t)k2 * CA];
      sPart[part][n] = s;
    }

    // --- WMMA phase 1: acc = relu(C_keys)@Wsm + rmsnorm(Z)@Wz ---
    v8f acc = {0.f, 0.f, 0.f, 0.f, 0.f, 0.f, 0.f, 0.f};
    const int arow = mtile * 16 + (lane & 15);
    const int bcol = ntile * 16 + (lane & 15);
#pragma unroll
    for (int kc = 0; kc < 4; ++kc) {
      v16bf a = load_a(sA1, CT, arow, kc, lane);
      v16bf b = load_b(sWTm, CT, bcol, kc, lane);
      acc = __builtin_amdgcn_wmma_f32_16x16x32_bf16(false, a, false, b,
                                                    (short)0, acc, false, false);
    }
#pragma unroll
    for (int kc = 0; kc < 4; ++kc) {
      v16bf a = load_a(sA2, CT, arow, kc, lane);
      v16bf b = load_b(sWTz, CT, bcol, kc, lane);
      acc = __builtin_amdgcn_wmma_f32_16x16x32_bf16(false, a, false, b,
                                                    (short)0, acc, false, false);
    }
    __syncthreads();   // sPart ready; A buffers free for reuse

    // --- geometry embed + single_l broadcast add ---
    {
      const int n  = ntile * 16 + (lane & 15);
      const int hi = lane >> 4;
      const float mcl = is_mc[l] ? 1.f : 0.f;
      const float msl = is_ms[l] ? 1.f : 0.f;
      const float wdm0 = sGeoW[n],       wdm1 = sGeoW[64 + n],  wdm2 = sGeoW[128 + n];
      const float wim  = sGeoW[192 + n], wmm  = sGeoW[256 + n];
      const float wdr0 = sGeoW[320 + n], wdr1 = sGeoW[384 + n], wdr2 = sGeoW[448 + n];
      const float wir  = sGeoW[512 + n], wmr  = sGeoW[576 + n];
      const float slv = sPart[0][n] + sPart[1][n] + sPart[2][n] + sPart[3][n];
#pragma unroll
      for (int r = 0; r < 8; ++r) {
        const int m = mtile * 16 + hi * 8 + r;
        const float* rs = &sRow[m * 12];
        const float gm = rs[4] * (rs[0] * wdm0 + rs[1] * wdm1 + rs[2] * wdm2 +
                                  rs[3] * wim + rs[4] * wmm);
        const float gr = rs[9] * (rs[5] * wdr0 + rs[6] * wdr1 + rs[7] * wdr2 +
                                  rs[8] * wir + rs[9] * wmr);
        acc[r] += mcl * gm + msl * gr + slv;
      }
    }

    // --- 3-layer MLP via LDS round-trip, keep P in registers ---
    const v8f p = acc;
    v8f cur = acc;
    const ushort_t* mw[3] = {sWT1, sWT2, sWT3};
#pragma unroll
    for (int layer = 0; layer < 3; ++layer) {
      __syncthreads();
      {
        const int n  = ntile * 16 + (lane & 15);
        const int hi = lane >> 4;
#pragma unroll
        for (int r = 0; r < 8; ++r) {
          const int m = mtile * 16 + hi * 8 + r;
          sH[m * CA + n] = f2bf(fmaxf(cur[r], 0.f));
        }
      }
      __syncthreads();
      v8f h = {0.f, 0.f, 0.f, 0.f, 0.f, 0.f, 0.f, 0.f};
#pragma unroll
      for (int kc = 0; kc < 2; ++kc) {
        v16bf a = load_a(sH, CA, arow, kc, lane);
        v16bf b = load_b(mw[layer], CA, bcol, kc, lane);
        h = __builtin_amdgcn_wmma_f32_16x16x32_bf16(false, a, false, b,
                                                    (short)0, h, false, false);
      }
      cur = h;
    }

    // --- out = P + h3, staged through LDS for coalesced float4 stores ---
    __syncthreads();
    {
      const int n  = ntile * 16 + (lane & 15);
      const int hi = lane >> 4;
#pragma unroll
      for (int r = 0; r < 8; ++r) {
        const int m = mtile * 16 + hi * 8 + r;
        sPout[m * CA + n] = p[r] + cur[r];
      }
    }
    __syncthreads();
    {
      float4* dst = (float4*)(out + ((size_t)d * Lq + l) * Kq * CA);
      const float4* s4 = (const float4*)sPout;
      for (int i = tid; i < Kq * CA / 4; i += 256) dst[i] = s4[i];
    }
    __syncthreads();
  }
}

extern "C" void kernel_launch(void* const* d_in, const int* in_sizes, int n_in,
                              void* d_out, int out_size, void* d_ws, size_t ws_size,
                              hipStream_t stream) {
  (void)in_sizes; (void)n_in; (void)out_size; (void)ws_size;
  const int*   indices   = (const int*)d_in[0];
  const float* C_L       = (const float*)d_in[1];
  const float* Z         = (const float*)d_in[2];
  const int*   tok_idx   = (const int*)d_in[3];
  const float* motif_pos = (const float*)d_in[4];
  const float* ref_pos   = (const float*)d_in[5];
  const int*   ref_uid   = (const int*)d_in[6];
  const unsigned char* is_mc = (const unsigned char*)d_in[7];
  const unsigned char* is_ms = (const unsigned char*)d_in[8];
  const float* Wdm  = (const float*)d_in[9];
  const float* Wim  = (const float*)d_in[10];
  const float* Wmm  = (const float*)d_in[11];
  const float* Wdr  = (const float*)d_in[12];
  const float* Wir  = (const float*)d_in[13];
  const float* Wmr  = (const float*)d_in[14];
  const float* Wsl  = (const float*)d_in[15];
  const float* Wsm  = (const float*)d_in[16];
  const float* rmsw = (const float*)d_in[17];
  const float* Wz   = (const float*)d_in[18];
  const float* W1   = (const float*)d_in[19];
  const float* W2   = (const float*)d_in[20];
  const float* W3   = (const float*)d_in[21];
  ushort_t* wt = (ushort_t*)d_ws;   // 57344 bytes of bf16 transposed weights

  prep_weights<<<dim3(112), dim3(256), 0, stream>>>(Wsm, Wz, W1, W2, W3, wt);
  fused_kernel<<<dim3(Dq * (Lq / NL)), dim3(256), 0, stream>>>(
      indices, C_L, Z, tok_idx, motif_pos, ref_pos, ref_uid, is_mc, is_ms,
      Wdm, Wim, Wmm, Wdr, Wir, Wmr, Wsl, rmsw, wt, (float*)d_out);
}
